// VectorQuantizer_89146341196193
// MI455X (gfx1250) — matrix operand order repaired and checked by reading, not compiled, not run on hardware
//
#include <hip/hip_runtime.h>
#include <hip/hip_bf16.h>

// ---------------------------------------------------------------------------
// VectorQuantizer for MI455X (gfx1250), wave32, bf16 WMMA + async-LDS path.
//
//   dots = tokens(32768x256) @ embed^T(256x2048)   -> 34.4 GFLOP (dominant)
//   idx  = argmin_k (||e_k||^2 - 2*dots)           -> fused, dots never stored
//   out  = embed[idx] back to NCHW, plus idx (as float) and loss=0
//
// Roofline: ~70MB HBM (~3us floor) vs 34.4 GFLOP -> compute bound; use
// v_wmma_f32_16x16x32_bf16. Codebook converted to bf16 once (1MB, L2-resident)
// and streamed into double-buffered LDS with global_load_async_to_lds_b128
// (ASYNCcnt) so chunk fills overlap the WMMA stream. Tile loop fully unrolled
// -> 4 independent WMMA accumulation chains per chunk.
// ---------------------------------------------------------------------------

typedef __attribute__((ext_vector_type(16))) __bf16 v16bf;
typedef __attribute__((ext_vector_type(8)))  float  v8f;

#define C_DIM       256
#define K_CODES     2048
#define HW_DIM      1024
#define BATCH       32
#define NTOK        (BATCH * HW_DIM)         // 32768 tokens
#define QELEMS      (BATCH * C_DIM * HW_DIM) // 8388608 output elems
#define TOK_PER_BLK 128
#define CODE_CHUNK  64
#define NCHUNK      (K_CODES / CODE_CHUNK)   // 32
#define LDS_STRIDE  264                      // 256 + 8 pad (ushort) vs bank conflicts
#define ROW_BYTES   (LDS_STRIDE * 2)         // 528, 16B-aligned

// LDS layout (dynamic shared, base offset 0)
#define LTOK_OFF    0
#define LTOK_BYTES  (TOK_PER_BLK * ROW_BYTES)            // 67584
#define EMBBUF_BYTES (CODE_CHUNK * ROW_BYTES)            // 33792
#define EMB0_OFF    LTOK_BYTES
#define EMB1_OFF    (LTOK_BYTES + EMBBUF_BYTES)
#define LDS_TOTAL   (LTOK_BYTES + 2 * EMBBUF_BYTES)      // 135168

union V16U { v16bf v; uint4 u[2]; };

__device__ __forceinline__ unsigned short f2bf(float f) {
    unsigned int u = __float_as_uint(f);
    u += 0x7FFFu + ((u >> 16) & 1u);          // round-to-nearest-even
    return (unsigned short)(u >> 16);
}

// ---------------------------------------------------------------------------
// Prep kernel A: ||e_k||^2 per code.
// ---------------------------------------------------------------------------
__global__ void __launch_bounds__(256)
vq_esq_kernel(const float* __restrict__ emb, float* __restrict__ esq) {
    int k = blockIdx.x * blockDim.x + threadIdx.x;
    if (k >= K_CODES) return;
    const float4* p = (const float4*)(emb + (size_t)k * C_DIM);
    float s = 0.f;
    #pragma unroll 8
    for (int i = 0; i < C_DIM / 4; ++i) {
        float4 q = p[i];
        s += q.x * q.x + q.y * q.y + q.z * q.z + q.w * q.w;
    }
    esq[k] = s;
}

// ---------------------------------------------------------------------------
// Prep kernel B: one-time f32 -> bf16 codebook conversion (1MB, L2-resident).
// ---------------------------------------------------------------------------
__global__ void __launch_bounds__(256)
vq_cvt_kernel(const float* __restrict__ emb, unsigned short* __restrict__ embb) {
    int i = blockIdx.x * blockDim.x + threadIdx.x;  // over K*C/4 float4
    float4 q = ((const float4*)emb)[i];
    uint2 p;
    p.x = (unsigned)f2bf(q.x) | ((unsigned)f2bf(q.y) << 16);
    p.y = (unsigned)f2bf(q.z) | ((unsigned)f2bf(q.w) << 16);
    ((uint2*)embb)[i] = p;
}

// ---------------------------------------------------------------------------
// Async prefetch of one 64-code bf16 chunk into an LDS buffer.
// One global_load_async_to_lds_b128 moves 32 lanes x 16B = 512B = one code
// row; each of the 8 waves issues 8 rows. Tracked by ASYNCcnt (in-order).
// ---------------------------------------------------------------------------
__device__ __forceinline__ void prefetch_chunk(const unsigned short* __restrict__ gemb,
                                               int chunk, unsigned lds_base,
                                               int wave, int lane) {
    #pragma unroll
    for (int j = 0; j < 8; ++j) {
        const int row      = wave * 8 + j;
        const unsigned dst = lds_base + (unsigned)(row * ROW_BYTES + lane * 16);
        const int      off = chunk * (CODE_CHUNK * C_DIM * 2) + row * (C_DIM * 2) + lane * 16;
        asm volatile("global_load_async_to_lds_b128 %0, %1, %2 offset:0"
                     :: "v"(dst), "v"(off), "s"(gemb) : "memory");
    }
}

// ---------------------------------------------------------------------------
// Main kernel: fused bf16-WMMA GEMM + argmin.
//   256 threads = 8 waves; 128 tokens/block (16/wave, A-frags in registers);
//   codebook streamed through double-buffered LDS via async loads.
// ---------------------------------------------------------------------------
__global__ void __launch_bounds__(256)
vq_argmin_kernel(const float* __restrict__ x, const unsigned short* __restrict__ embb,
                 const float* __restrict__ esq, int* __restrict__ best,
                 float* __restrict__ out_idx)
{
    extern __shared__ unsigned short smem[];

    const int tid  = threadIdx.x;
    const int lane = tid & 31;
    const int wave = tid >> 5;
    const int hi   = lane >> 4;
    const int l15  = lane & 15;

    const int tok0 = blockIdx.x * TOK_PER_BLK;  // 128 | 1024 => single batch b
    const int b    = tok0 >> 10;
    const int hw0  = tok0 & (HW_DIM - 1);

    // kick off chunk 0 while we stage tokens
    prefetch_chunk(embb, 0, EMB0_OFF, wave, lane);

    // ---- NHWC-style token load: for fixed c, 128 consecutive hw contiguous
    {
        const int t     = tid & 127;
        const int c_off = tid >> 7;
        const float* xb = x + (size_t)b * C_DIM * HW_DIM + hw0;
        for (int c = c_off; c < C_DIM; c += 2) {
            smem[t * LDS_STRIDE + c] = f2bf(xb[(size_t)c * HW_DIM + t]);
        }
    }
    __syncthreads();

    // ---- preload this wave's A fragments (ISA 16-bit A 16x32 layout)
    v16bf afrag[8];
    {
        const unsigned short* rowp = &smem[(wave * 16 + l15) * LDS_STRIDE];
        #pragma unroll
        for (int ci = 0; ci < 8; ++ci) {
            V16U t;
            const int c0 = ci * 32 + hi * 8;
            t.u[0] = *(const uint4*)(rowp + c0);
            t.u[1] = *(const uint4*)(rowp + c0 + 16);
            afrag[ci] = t.v;
        }
    }

    float minv[8];
    int   mini[8];
    #pragma unroll
    for (int m = 0; m < 8; ++m) { minv[m] = 3.4e38f; mini[m] = 0; }

    for (int chunk = 0; chunk < NCHUNK; ++chunk) {
        const unsigned cur_off = (chunk & 1) ? EMB1_OFF : EMB0_OFF;
        // prefetch next chunk into the other buffer (prev compute on it is
        // fenced by the trailing barrier of the previous iteration)
        if (chunk + 1 < NCHUNK) {
            prefetch_chunk(embb, chunk + 1, (chunk & 1) ? EMB0_OFF : EMB1_OFF, wave, lane);
            asm volatile("s_wait_asynccnt 0x8" ::: "memory");  // current 8 rows landed
        } else {
            asm volatile("s_wait_asynccnt 0x0" ::: "memory");
        }
        __syncthreads();  // all waves' rows landed

        const unsigned short* bbase = smem + cur_off / 2;
        float es[4];
        #pragma unroll
        for (int ct = 0; ct < 4; ++ct)
            es[ct] = esq[chunk * CODE_CHUNK + ct * 16 + l15];

        // 4 code tiles of 16, fully unrolled: 4 independent WMMA accumulation
        // chains (32 static v_wmma per chunk body), statically selected es[ct].
        #pragma unroll
        for (int ct = 0; ct < 4; ++ct) {
            v8f acc = {};
            // B 32x16 layout: lanes 0-15 hold K 0..7,16..23, lanes 16-31 hold K 8..15,24..31
            const unsigned short* browp = bbase + (ct * 16 + l15) * LDS_STRIDE;
            #pragma unroll
            for (int ci = 0; ci < 8; ++ci) {
                V16U t;
                const int c0 = ci * 32 + hi * 8;
                t.u[0] = *(const uint4*)(browp + c0);
                t.u[1] = *(const uint4*)(browp + c0 + 16);
                acc = __builtin_amdgcn_wmma_f32_16x16x32_bf16(
                          false, afrag[ci], false, t.v, (short)0, acc, false, false);
            }
            const int code = chunk * CODE_CHUNK + ct * 16 + l15;
            // D layout: vgpr m, lane l -> token M = m + 8*(l>>4), code N = l&15
            #pragma unroll
            for (int m = 0; m < 8; ++m) {
                float s = es[ct] - 2.0f * acc[m];
                if (s < minv[m]) { minv[m] = s; mini[m] = code; }
            }
        }
        __syncthreads();  // everyone done with cur buffer before it is refilled
    }

    // ---- reduce (min,idx) across the 16-lane half holding each token's codes
    #pragma unroll
    for (int m = 0; m < 8; ++m) {
        float bv = minv[m];
        int   bi = mini[m];
        #pragma unroll
        for (int off = 8; off >= 1; off >>= 1) {
            float ov = __shfl_xor(bv, off, 32);
            int   oi = __shfl_xor(bi, off, 32);
            if (ov < bv || (ov == bv && oi < bi)) { bv = ov; bi = oi; } // first-match argmin
        }
        if (l15 == 0) {
            int tok = tok0 + wave * 16 + m + 8 * hi;
            best[tok]    = bi;
            out_idx[tok] = (float)bi;
        }
    }
}

// ---------------------------------------------------------------------------
// Gather embed[idx] back to NCHW with coalesced float4 stores.
// ---------------------------------------------------------------------------
__global__ void __launch_bounds__(256)
vq_gather_kernel(const float* __restrict__ emb, const int* __restrict__ best,
                 float* __restrict__ outq, float* __restrict__ loss)
{
    const int lin = blockIdx.x * blockDim.x + threadIdx.x;   // over QELEMS/4
    if (lin == 0) loss[0] = 0.0f;
    const int hw4 = lin & 255;
    const int c   = (lin >> 8) & 255;
    const int b   = lin >> 16;
    const int tb  = b * HW_DIM + hw4 * 4;
    float4 o;
    o.x = emb[(size_t)best[tb + 0] * C_DIM + c];
    o.y = emb[(size_t)best[tb + 1] * C_DIM + c];
    o.z = emb[(size_t)best[tb + 2] * C_DIM + c];
    o.w = emb[(size_t)best[tb + 3] * C_DIM + c];
    ((float4*)outq)[lin] = o;
}

// ---------------------------------------------------------------------------
extern "C" void kernel_launch(void* const* d_in, const int* in_sizes, int n_in,
                              void* d_out, int out_size, void* d_ws, size_t ws_size,
                              hipStream_t stream) {
    const float* x   = (const float*)d_in[0];   // [32,256,32,32]
    const float* emb = (const float*)d_in[1];   // [2048,256]
    float* out = (float*)d_out;                 // [quantize | idx | loss]

    // workspace: esq f32[2048] | embb bf16[2048*256] | best i32[32768]
    float*          esq  = (float*)d_ws;
    unsigned short* embb = (unsigned short*)((char*)d_ws + K_CODES * sizeof(float));
    int*            best = (int*)((char*)embb + (size_t)K_CODES * C_DIM * sizeof(unsigned short));

    vq_esq_kernel<<<K_CODES / 256, 256, 0, stream>>>(emb, esq);
    vq_cvt_kernel<<<K_CODES * C_DIM / 4 / 256, 256, 0, stream>>>(emb, embb);

    vq_argmin_kernel<<<NTOK / TOK_PER_BLK, 256, LDS_TOTAL, stream>>>(
        x, embb, esq, best, out + QELEMS);

    vq_gather_kernel<<<QELEMS / 4 / 256, 256, 0, stream>>>(
        emb, best, out, out + QELEMS + NTOK);
}
